// DrugGCNEncoder_81088982548909
// MI455X (gfx1250) — compile-verified, matrix operand order checked by reading, not run on hardware
//
#include <hip/hip_runtime.h>

typedef _Float16 half_t;
typedef __attribute__((ext_vector_type(16))) _Float16 v16h;
typedef __attribute__((ext_vector_type(8)))  _Float16 v8h;
typedef __attribute__((ext_vector_type(8)))  float    v8f;

#define NN    131072
#define EE    524288
#define GG    4096
#define C_IN  78
#define C_MID 156
#define C_BIG 312
#define C_OUT 128

// ---------------------------------------------------------------------------
// f16 WMMA GEMM on pre-padded operands:
//   Xh : M x KPAD   f16 row-major (K zero-padded to KPAD, multiple of 32)
//   WTh: Npad x KPAD f16 row-major (= W^T, zero-padded)
//   Y  : M x N      f32 row-major (ldy = N), optional bias / relu
// One wave per 16x16 D tile. A fragment per lane = two contiguous b128 loads;
// B fragment identical pattern thanks to the W^T layout. KPAD templated so
// the K loop fully unrolls into a chain of v_wmma_f32_16x16x32_f16.
// ---------------------------------------------------------------------------
template <int KPAD>
__global__ __launch_bounds__(32)
void gemm_wmma_f16(const half_t* __restrict__ Xh, const half_t* __restrict__ WTh,
                   const float* __restrict__ bias, float* __restrict__ Y,
                   int M, int N, int do_relu)
{
    const int lane = threadIdx.x;
    const int half = lane >> 4;
    const int l16  = lane & 15;
    const int mbase = blockIdx.y * 16;
    const int nbase = blockIdx.x * 16;

    const half_t* __restrict__ ar = Xh  + (size_t)(mbase + l16) * KPAD + half * 8;
    const half_t* __restrict__ br = WTh + (size_t)(nbase + l16) * KPAD + half * 8;

    v8f acc = {};
#pragma unroll
    for (int k0 = 0; k0 < KPAD; k0 += 32) {
        const v8h a0 = *(const v8h*)(ar + k0);
        const v8h a1 = *(const v8h*)(ar + k0 + 16);
        const v8h b0 = *(const v8h*)(br + k0);
        const v8h b1 = *(const v8h*)(br + k0 + 16);
        const v16h a = __builtin_shufflevector(a0, a1, 0,1,2,3,4,5,6,7,8,9,10,11,12,13,14,15);
        const v16h b = __builtin_shufflevector(b0, b1, 0,1,2,3,4,5,6,7,8,9,10,11,12,13,14,15);
        // (neg_a, A, neg_b, B, c_mod, C, reuse_a, reuse_b)
        acc = __builtin_amdgcn_wmma_f32_16x16x32_f16(false, a, false, b,
                                                     (short)0, acc, false, false);
    }

    const int n = nbase + l16;
    if (n < N) {
        const float bv = bias ? bias[n] : 0.0f;
#pragma unroll
        for (int r = 0; r < 8; ++r) {
            const int m = mbase + r + half * 8;
            float v = acc[r] + bv;
            if (do_relu) v = fmaxf(v, 0.0f);
            Y[(size_t)m * (size_t)N + n] = v;
        }
    }
}

// ---------------------------------------------------------------------------
// Staging conversions (pad K to KPAD with zeros; weights also transposed)
// ---------------------------------------------------------------------------
__global__ void cvt_pad_rows(const float* __restrict__ X, half_t* __restrict__ Xh,
                             int M, int K, int ldx, int Kpad)
{
    size_t idx = (size_t)blockIdx.x * blockDim.x + threadIdx.x;
    size_t total = (size_t)M * (size_t)Kpad;
    if (idx >= total) return;
    int row = (int)(idx / Kpad);
    int k   = (int)(idx % Kpad);
    Xh[idx] = (k < K) ? (half_t)X[(size_t)row * ldx + k] : (half_t)0.0f;
}

__global__ void cvt_pad_wT(const float* __restrict__ W, half_t* __restrict__ WTh,
                           int K, int N, int Kpad, int Npad)
{
    size_t idx = (size_t)blockIdx.x * blockDim.x + threadIdx.x;
    size_t total = (size_t)Npad * (size_t)Kpad;
    if (idx >= total) return;
    int n = (int)(idx / Kpad);
    int k = (int)(idx % Kpad);
    WTh[idx] = (k < K && n < N) ? (half_t)W[(size_t)k * N + n] : (half_t)0.0f;
}

// ---------------------------------------------------------------------------
// Graph kernels
// ---------------------------------------------------------------------------
__global__ void fill_f32(float* __restrict__ p, float v, size_t n)
{
    size_t i = (size_t)blockIdx.x * blockDim.x + threadIdx.x;
    if (i < n) p[i] = v;
}

__global__ void degree_kernel(const int* __restrict__ dst, float* __restrict__ deg, int E)
{
    int e = blockIdx.x * blockDim.x + threadIdx.x;
    if (e < E) atomicAdd(&deg[dst[e]], 1.0f);
}

__global__ void dis_kernel(const float* __restrict__ deg, float* __restrict__ dis, int n)
{
    int i = blockIdx.x * blockDim.x + threadIdx.x;
    if (i < n) dis[i] = rsqrtf(fmaxf(deg[i], 1.0f));   // deg >= 1 (self-loop)
}

__global__ void norm_kernel(const int* __restrict__ src, const int* __restrict__ dst,
                            const float* __restrict__ dis, float* __restrict__ norm, int E)
{
    int e = blockIdx.x * blockDim.x + threadIdx.x;
    if (e < E) norm[e] = dis[src[e]] * dis[dst[e]];
}

// one block per edge; threads = channels (coalesced atomics on dst row)
__global__ void scatter_kernel(const float* __restrict__ T, const int* __restrict__ src,
                               const int* __restrict__ dst, const float* __restrict__ norm,
                               float* __restrict__ agg, int C)
{
    const int e = blockIdx.x;
    const int c = threadIdx.x;
    if (c >= C) return;
    const int s  = src[e];
    const int d  = dst[e];
    const float w = norm[e];
    atomicAdd(&agg[(size_t)d * (size_t)C + c], T[(size_t)s * (size_t)C + c] * w);
}

// agg[i,c] = relu( agg[i,c] + T[i,c]*dis[i]^2 + bias[c] )   (self-loop + bias)
__global__ void combine_kernel(const float* __restrict__ T, const float* __restrict__ dis,
                               const float* __restrict__ bias, float* __restrict__ agg,
                               int nodes, int C)
{
    size_t idx = (size_t)blockIdx.x * blockDim.x + threadIdx.x;
    size_t total = (size_t)nodes * (size_t)C;
    if (idx >= total) return;
    int node = (int)(idx / C);
    int c    = (int)(idx % C);
    float d  = dis[node];
    float v  = agg[idx] + T[idx] * d * d + bias[c];
    agg[idx] = fmaxf(v, 0.0f);
}

// batch = node/32 (32 contiguous nodes per graph): per-graph channel max
__global__ void pool_kernel(const float* __restrict__ h, float* __restrict__ pooled,
                            int nodes_per_g, int C)
{
    const int g = blockIdx.x;
    const int c = threadIdx.x;
    if (c >= C) return;
    const float* base = h + (size_t)g * nodes_per_g * C + c;
    float m = -3.0e38f;
    for (int i = 0; i < nodes_per_g; ++i)
        m = fmaxf(m, base[(size_t)i * C]);
    pooled[(size_t)g * C + c] = m;
}

// ---------------------------------------------------------------------------
static inline dim3 blocks1d(size_t total, int bs) { return dim3((unsigned)((total + bs - 1) / bs)); }

extern "C" void kernel_launch(void* const* d_in, const int* in_sizes, int n_in,
                              void* d_out, int out_size, void* d_ws, size_t ws_size,
                              hipStream_t stream)
{
    (void)in_sizes; (void)n_in; (void)out_size; (void)ws_size;

    const float* x    = (const float*)d_in[0];
    const int*   ei   = (const int*)  d_in[1];
    const float* W1   = (const float*)d_in[3];
    const float* b1   = (const float*)d_in[4];
    const float* W2   = (const float*)d_in[5];
    const float* b2   = (const float*)d_in[6];
    const float* W3   = (const float*)d_in[7];
    const float* b3   = (const float*)d_in[8];
    const float* fcW  = (const float*)d_in[9];
    const float* fcb  = (const float*)d_in[10];
    float* out = (float*)d_out;

    const int* src = ei;        // edge_index[0]
    const int* dst = ei + EE;   // edge_index[1]

    // -------- workspace layout --------
    float* ws = (float*)d_ws;
    size_t off = 0;                         // in floats
    float* deg  = ws + off; off += NN;
    float* dis  = ws + off; off += NN;
    float* nrm  = ws + off; off += EE;
    const size_t BIG = (size_t)NN * (size_t)C_BIG;
    float* bufA = ws + off; off += BIG;     // linear-transform output T each layer
    float* bufB = ws + off; off += BIG;     // agg L1 / agg L3 (h1, h3)
    float* bufC = ws + off; off += BIG;     // agg L2 / pooled (h2)
    half_t* Hh  = (half_t*)(ws + off); off += (size_t)NN * 160 / 2;  // f16 activations (max N x 160)
    half_t* Wh  = (half_t*)(ws + off); off += (320 * 320) / 2;       // f16 transposed weights

    // -------- gcn_norm --------
    fill_f32<<<blocks1d(NN, 256), 256, 0, stream>>>(deg, 1.0f, (size_t)NN);
    degree_kernel<<<blocks1d(EE, 256), 256, 0, stream>>>(dst, deg, EE);
    dis_kernel<<<blocks1d(NN, 256), 256, 0, stream>>>(deg, dis, NN);
    norm_kernel<<<blocks1d(EE, 256), 256, 0, stream>>>(src, dst, dis, nrm, EE);

    // -------- layer 1: 78 -> 78 (Kpad 96, Npad 80) --------
    {
        size_t xh = (size_t)NN * 96;
        cvt_pad_rows<<<blocks1d(xh, 256), 256, 0, stream>>>(x, Hh, NN, C_IN, C_IN, 96);
        size_t wh = (size_t)80 * 96;
        cvt_pad_wT<<<blocks1d(wh, 256), 256, 0, stream>>>(W1, Wh, C_IN, C_IN, 96, 80);
        dim3 grid((C_IN + 15) / 16, NN / 16);
        gemm_wmma_f16<96><<<grid, 32, 0, stream>>>(Hh, Wh, nullptr, bufA, NN, C_IN, 0);

        size_t tot = (size_t)NN * C_IN;
        fill_f32<<<blocks1d(tot, 256), 256, 0, stream>>>(bufB, 0.0f, tot);
        scatter_kernel<<<EE, 128, 0, stream>>>(bufA, src, dst, nrm, bufB, C_IN);
        combine_kernel<<<blocks1d(tot, 256), 256, 0, stream>>>(bufA, dis, b1, bufB, NN, C_IN);
    }
    // -------- layer 2: 78 -> 156 (Kpad 96, Npad 160) --------
    {
        size_t xh = (size_t)NN * 96;
        cvt_pad_rows<<<blocks1d(xh, 256), 256, 0, stream>>>(bufB, Hh, NN, C_IN, C_IN, 96);
        size_t wh = (size_t)160 * 96;
        cvt_pad_wT<<<blocks1d(wh, 256), 256, 0, stream>>>(W2, Wh, C_IN, C_MID, 96, 160);
        dim3 grid((C_MID + 15) / 16, NN / 16);
        gemm_wmma_f16<96><<<grid, 32, 0, stream>>>(Hh, Wh, nullptr, bufA, NN, C_MID, 0);

        size_t tot = (size_t)NN * C_MID;
        fill_f32<<<blocks1d(tot, 256), 256, 0, stream>>>(bufC, 0.0f, tot);
        scatter_kernel<<<EE, 192, 0, stream>>>(bufA, src, dst, nrm, bufC, C_MID);
        combine_kernel<<<blocks1d(tot, 256), 256, 0, stream>>>(bufA, dis, b2, bufC, NN, C_MID);
    }
    // -------- layer 3: 156 -> 312 (Kpad 160, Npad 320) --------
    {
        size_t xh = (size_t)NN * 160;
        cvt_pad_rows<<<blocks1d(xh, 256), 256, 0, stream>>>(bufC, Hh, NN, C_MID, C_MID, 160);
        size_t wh = (size_t)320 * 160;
        cvt_pad_wT<<<blocks1d(wh, 256), 256, 0, stream>>>(W3, Wh, C_MID, C_BIG, 160, 320);
        dim3 grid((C_BIG + 15) / 16, NN / 16);
        gemm_wmma_f16<160><<<grid, 32, 0, stream>>>(Hh, Wh, nullptr, bufA, NN, C_BIG, 0);

        size_t tot = (size_t)NN * C_BIG;
        fill_f32<<<blocks1d(tot, 256), 256, 0, stream>>>(bufB, 0.0f, tot);
        scatter_kernel<<<EE, 320, 0, stream>>>(bufA, src, dst, nrm, bufB, C_BIG);
        combine_kernel<<<blocks1d(tot, 256), 256, 0, stream>>>(bufA, dis, b3, bufB, NN, C_BIG);
    }
    // -------- global max pool (32 contiguous nodes per graph) --------
    pool_kernel<<<GG, 320, 0, stream>>>(bufB, bufC, NN / GG, C_BIG);

    // -------- FC head: 312 -> 128 (Kpad 320), +bias, no relu --------
    {
        size_t xh = (size_t)GG * 320;
        cvt_pad_rows<<<blocks1d(xh, 256), 256, 0, stream>>>(bufC, Hh, GG, C_BIG, C_BIG, 320);
        size_t wh = (size_t)128 * 320;
        cvt_pad_wT<<<blocks1d(wh, 256), 256, 0, stream>>>(fcW, Wh, C_BIG, C_OUT, 320, 128);
        dim3 grid(C_OUT / 16, GG / 16);
        gemm_wmma_f16<320><<<grid, 32, 0, stream>>>(Hh, Wh, fcb, out, GG, C_OUT, 0);
    }
}
